// OccupancyDecoder_85109071938165
// MI455X (gfx1250) — matrix operand order repaired and checked by reading, not compiled, not run on hardware
//
#include <hip/hip_runtime.h>

// ---------------- problem constants ----------------
#define DDIM   64
#define DHW    (64*64*64)          // 262144
#define CCH    128                 // input feature channels
#define H1DIM  512
#define H2DIM  512
#define NPTS   65536
#define NBATCH 2
#define MTOT   (NBATCH*NPTS)       // 131072 rows
#define TILE_M 128                 // rows per workgroup

// LDS layout (halves), padded for bank-conflict-free 16B fragment loads
#define AF_STRIDE 136              // 128 + 8 pad -> row bank offset = 4*row
#define H1_STRIDE 520              // 512 + 8 pad -> row bank offset = 4*row
#define AF_BYTES  (TILE_M*AF_STRIDE*2)   // 34816
#define H1_BYTES  (TILE_M*H1_STRIDE*2)   // 133120
#define SMEM_BYTES (AF_BYTES + H1_BYTES) // 167936  (< 320KB WGP LDS)

// ws layout (bytes)
#define WS_W1T 0                           // 512*128 f16 = 131072 B
#define WS_W2T (WS_W1T + CCH*H1DIM*2)      // 512*512 f16 = 524288 B
#define WS_A1  (WS_W2T + H1DIM*H2DIM*2)    // 512 f32
#define WS_C1  (WS_A1 + H1DIM*4)
#define WS_A2  (WS_C1 + H1DIM*4)
#define WS_C2  (WS_A2 + H2DIM*4)

typedef _Float16 h8  __attribute__((ext_vector_type(8)));
typedef _Float16 h16 __attribute__((ext_vector_type(16)));
typedef float    v8f __attribute__((ext_vector_type(8)));

__device__ __forceinline__ h16 frag_cat(h8 lo, h8 hi) {
    return __builtin_shufflevector(lo, hi, 0,1,2,3,4,5,6,7,8,9,10,11,12,13,14,15);
}

// ---------------- prep: transpose weights to f16, fold BN constants ----------------
__global__ void occ_prep(const float* __restrict__ W1, const float* __restrict__ W2,
                         const float* __restrict__ b1, const float* __restrict__ g1,
                         const float* __restrict__ be1, const float* __restrict__ m1,
                         const float* __restrict__ v1,
                         const float* __restrict__ b2, const float* __restrict__ g2,
                         const float* __restrict__ be2, const float* __restrict__ m2,
                         const float* __restrict__ v2,
                         _Float16* __restrict__ W1t, _Float16* __restrict__ W2t,
                         float* __restrict__ a1, float* __restrict__ c1,
                         float* __restrict__ a2, float* __restrict__ c2)
{
    int idx = blockIdx.x * blockDim.x + threadIdx.x;
    if (idx < H1DIM * H2DIM) {                 // W2 is (512,512): W2[k][n] -> W2t[n][k]
        int k = idx / H2DIM, n = idx % H2DIM;
        W2t[n * H1DIM + k] = (_Float16)W2[idx];
    }
    if (idx < CCH * H1DIM) {                   // W1 is (128,512): W1[c][h] -> W1t[h][c]
        int c = idx / H1DIM, h = idx % H1DIM;
        W1t[h * CCH + c] = (_Float16)W1[idx];
    }
    if (idx < H1DIM) {
        float rs1 = rsqrtf(v1[idx] + 1e-5f) * g1[idx];
        a1[idx] = rs1;
        c1[idx] = (b1[idx] - m1[idx]) * rs1 + be1[idx];
        float rs2 = rsqrtf(v2[idx] + 1e-5f) * g2[idx];
        a2[idx] = rs2;
        c2[idx] = (b2[idx] - m2[idx]) * rs2 + be2[idx];
    }
}

// ---------------- fused megakernel: sample -> GEMM1 -> GEMM2(+GEMM3) ----------------
__global__ void __launch_bounds__(256, 1)
occ_main(const float* __restrict__ feat, const float* __restrict__ query,
         const _Float16* __restrict__ W1t, const _Float16* __restrict__ W2t,
         const float* __restrict__ a1, const float* __restrict__ c1,
         const float* __restrict__ a2, const float* __restrict__ c2,
         const float* __restrict__ W3, const float* __restrict__ b3,
         float* __restrict__ out)
{
    extern __shared__ char smem[];
    _Float16* Af = (_Float16*)smem;                 // 128 x (128+8) f16 sampled feats
    _Float16* H1 = (_Float16*)(smem + AF_BYTES);    // 128 x (512+8) f16 hidden1

    const int tid  = threadIdx.x;
    const int lane = tid & 31;          // wave32
    const int wv   = tid >> 5;          // 8 waves
    const int hi   = lane >> 4;         // half-wave group
    const int ln   = lane & 15;
    const int mbase = blockIdx.x * TILE_M;

    // ============ Stage 0: trilinear sampling (border pad, align_corners) ============
    {
        const int p    = tid >> 1;            // point 0..127 in tile
        const int csel = (tid & 1) * 64;      // channel half handled by this thread
        const int m = mbase + p;
        const int b = m >> 16;                // m / 65536
        const int n = m & (NPTS - 1);
        const float* q = query + ((size_t)b * NPTS + n) * 3;
        // grid = 2q-1; unnorm = (grid+1)*0.5*(dim-1) = q*(dim-1); border = clamp
        float x = fminf(fmaxf(q[0] * 63.f, 0.f), 63.f);
        float y = fminf(fmaxf(q[1] * 63.f, 0.f), 63.f);
        float z = fminf(fmaxf(q[2] * 63.f, 0.f), 63.f);
        int x0 = (int)floorf(x); int x1 = min(x0 + 1, 63);
        int y0 = (int)floorf(y); int y1 = min(y0 + 1, 63);
        int z0 = (int)floorf(z); int z1 = min(z0 + 1, 63);
        float wx = x - (float)x0, wy = y - (float)y0, wz = z - (float)z0;
        float ux = 1.f - wx, uy = 1.f - wy, uz = 1.f - wz;

        int i000 = (z0 * DDIM + y0) * DDIM + x0;
        int i001 = (z0 * DDIM + y0) * DDIM + x1;
        int i010 = (z0 * DDIM + y1) * DDIM + x0;
        int i011 = (z0 * DDIM + y1) * DDIM + x1;
        int i100 = (z1 * DDIM + y0) * DDIM + x0;
        int i101 = (z1 * DDIM + y0) * DDIM + x1;
        int i110 = (z1 * DDIM + y1) * DDIM + x0;
        int i111 = (z1 * DDIM + y1) * DDIM + x1;
        float w000 = uz*uy*ux, w001 = uz*uy*wx, w010 = uz*wy*ux, w011 = uz*wy*wx;
        float w100 = wz*uy*ux, w101 = wz*uy*wx, w110 = wz*wy*ux, w111 = wz*wy*wx;

        const float* base = feat + ((size_t)b * CCH + csel) * (size_t)DHW;
        _Float16* dst = Af + p * AF_STRIDE + csel;
        #pragma unroll 4
        for (int c = 0; c < 64; ++c) {
            const float* fp = base + (size_t)c * DHW;
            float v = w000 * fp[i000] + w001 * fp[i001]
                    + w010 * fp[i010] + w011 * fp[i011]
                    + w100 * fp[i100] + w101 * fp[i101]
                    + w110 * fp[i110] + w111 * fp[i111];
            dst[c] = (_Float16)v;
        }
    }
    __syncthreads();

    // ============ Stage 1: GEMM1 (128 -> 512) + bias/BN/ReLU -> H1 (LDS, f16) ============
    // wave wv owns rows [wv*16, wv*16+16). A frag (ISA 16-bit A 16x32): lane row = ln,
// half i -> K = (i&7) + 8*hi + 16*(i>>3).  B frag: lane col = ln, same K order as A.
    h16 afr1[4];
    {
        const _Float16* arow = Af + (wv * 16 + ln) * AF_STRIDE;
        #pragma unroll
        for (int kb = 0; kb < 4; ++kb) {
            int k0 = kb * 32 + hi * 8;
            afr1[kb] = frag_cat(*(const h8*)(arow + k0), *(const h8*)(arow + k0 + 16));
        }
    }
    for (int cb = 0; cb < 32; ++cb) {
        const int ncol = cb * 16 + ln;
        const _Float16* brow = W1t + (size_t)ncol * CCH;   // W1t[n][k], contiguous K
        v8f acc = {};
        #pragma unroll
        for (int kb = 0; kb < 4; ++kb) {
            int k0 = kb * 32 + hi * 8;
            h16 bfr = frag_cat(*(const h8*)(brow + k0), *(const h8*)(brow + k0 + 16));
            acc = __builtin_amdgcn_wmma_f32_16x16x32_f16(
                      false, afr1[kb], false, bfr, (short)0, acc, false, false);
        }
        const float sa = a1[ncol], sc = c1[ncol];
        _Float16* hcol = H1 + cb * 16 + ln;
        #pragma unroll
        for (int r = 0; r < 8; ++r) {                     // D: reg r -> M = r + 8*hi
            float hv = fmaxf(acc[r] * sa + sc, 0.f);
            hcol[(wv * 16 + r + 8 * hi) * H1_STRIDE] = (_Float16)hv;
        }
    }
    __syncthreads();

    // ============ Stage 2+3: GEMM2 (512 -> 512) + BN/ReLU, fused dot with W3 ============
    float opart[8] = {0.f, 0.f, 0.f, 0.f, 0.f, 0.f, 0.f, 0.f};
    const _Float16* arow2 = H1 + (wv * 16 + ln) * H1_STRIDE;
    for (int cb = 0; cb < 32; ++cb) {
        const int ncol = cb * 16 + ln;
        const _Float16* brow = W2t + (size_t)ncol * H1DIM;
        v8f acc = {};
        #pragma unroll
        for (int kb = 0; kb < 16; ++kb) {
            int k0a = kb * 32 + hi * 8;
            h16 afr = frag_cat(*(const h8*)(arow2 + k0a), *(const h8*)(arow2 + k0a + 16));
            int k0b = kb * 32 + hi * 8;
            h16 bfr = frag_cat(*(const h8*)(brow + k0b), *(const h8*)(brow + k0b + 16));
            acc = __builtin_amdgcn_wmma_f32_16x16x32_f16(
                      false, afr, false, bfr, (short)0, acc, false, false);
        }
        const float sa = a2[ncol], sc = c2[ncol], w3v = W3[ncol];
        #pragma unroll
        for (int r = 0; r < 8; ++r) {
            float hv = fmaxf(acc[r] * sa + sc, 0.f);
            opart[r] += hv * w3v;   // GEMM3 folded into epilogue: out[row] += h2*W3[n]
        }
    }
    // reduce partial sums across the 16 lanes that share each M (xor of bits 0..3
    // stays inside each half-wave group)
    const float bb = b3[0];
    #pragma unroll
    for (int r = 0; r < 8; ++r) {
        float v = opart[r];
        v += __shfl_xor(v, 1, 32);
        v += __shfl_xor(v, 2, 32);
        v += __shfl_xor(v, 4, 32);
        v += __shfl_xor(v, 8, 32);
        if (ln == 0)
            out[mbase + wv * 16 + r + 8 * hi] = v + bb;
    }
}

// ---------------- host launcher ----------------
extern "C" void kernel_launch(void* const* d_in, const int* in_sizes, int n_in,
                              void* d_out, int out_size, void* d_ws, size_t ws_size,
                              hipStream_t stream) {
    (void)in_sizes; (void)n_in; (void)out_size; (void)ws_size;
    const float* feat = (const float*)d_in[0];
    const float* qry  = (const float*)d_in[1];
    const float* W1   = (const float*)d_in[2];
    const float* b1   = (const float*)d_in[3];
    const float* g1   = (const float*)d_in[4];
    const float* be1  = (const float*)d_in[5];
    const float* m1   = (const float*)d_in[6];
    const float* v1   = (const float*)d_in[7];
    const float* W2   = (const float*)d_in[8];
    const float* b2   = (const float*)d_in[9];
    const float* g2   = (const float*)d_in[10];
    const float* be2  = (const float*)d_in[11];
    const float* m2   = (const float*)d_in[12];
    const float* v2   = (const float*)d_in[13];
    const float* W3   = (const float*)d_in[14];
    const float* b3   = (const float*)d_in[15];
    float* out = (float*)d_out;

    char* ws = (char*)d_ws;
    _Float16* W1t = (_Float16*)(ws + WS_W1T);
    _Float16* W2t = (_Float16*)(ws + WS_W2T);
    float* a1 = (float*)(ws + WS_A1);
    float* c1 = (float*)(ws + WS_C1);
    float* a2 = (float*)(ws + WS_A2);
    float* c2 = (float*)(ws + WS_C2);

    occ_prep<<<(H1DIM * H2DIM + 255) / 256, 256, 0, stream>>>(
        W1, W2, b1, g1, be1, m1, v1, b2, g2, be2, m2, v2,
        W1t, W2t, a1, c1, a2, c2);

    (void)hipFuncSetAttribute((const void*)occ_main,
                              hipFuncAttributeMaxDynamicSharedMemorySize, SMEM_BYTES);
    occ_main<<<MTOT / TILE_M, 256, SMEM_BYTES, stream>>>(
        feat, qry, W1t, W2t, a1, c1, a2, c2, W3, b3, out);
}
